// Attention_layer_41755672051886
// MI455X (gfx1250) — matrix operand order repaired and checked by reading, not compiled, not run on hardware
//
#include <hip/hip_runtime.h>
#include <hip/hip_bf16.h>
#include <math.h>

typedef __bf16 bf16_t;
typedef __attribute__((ext_vector_type(16))) __bf16 v16bf;
typedef __attribute__((ext_vector_type(8)))  __bf16 v8bf;
typedef __attribute__((ext_vector_type(8)))  float  v8f;
typedef __attribute__((ext_vector_type(4)))  unsigned int u32x4;
typedef __attribute__((ext_vector_type(8)))  int i32x8;
typedef __attribute__((ext_vector_type(4)))  int i32x4;

#define SROWF 1032  // f32 stride per score row in LDS (1024 + 8 pad)

// Build a 16x32 bf16 A-fragment for lane: elems 0..15 at p[0..15]
__device__ __forceinline__ v16bf make_a_frag(const bf16_t* p) {
    v8bf lo = *(const v8bf*)p;
    v8bf hi = *(const v8bf*)(p + 8);
    v16bf r;
#pragma unroll
    for (int i = 0; i < 8; ++i) { r[i] = lo[i]; r[i + 8] = hi[i]; }
    return r;
}

// 8 elements per thread; all sizes are multiples of 8
__global__ void cvt_f32_to_bf16(const float* __restrict__ src, bf16_t* __restrict__ dst, long n) {
    long i = ((long)blockIdx.x * blockDim.x + threadIdx.x) * 8;
    if (i + 8 <= n) {
        v8bf o;
#pragma unroll
        for (int j = 0; j < 8; ++j) o[j] = (bf16_t)src[i + j];
        *(v8bf*)(dst + i) = o;
    }
}

// C = A[MxK] @ W[NxK]^T + bias   (A, W bf16 row-major; K % 32 == 0, N % 64 == 0)
// OUT_MODE 1: bf16 row-major   2: bf16 transposed C[n][m]
template <int OUT_MODE>
__global__ __launch_bounds__(128) void gemm_xwT(
    const bf16_t* __restrict__ A, int lda,
    const bf16_t* __restrict__ W, int ldw,
    const float* __restrict__ bias,
    void* __restrict__ Cout, int ldc,
    int M, int N, int K)
{
    const int lane = threadIdx.x & 31;
    const int wave = threadIdx.x >> 5;
    const int half = lane >> 4;
    const int l16  = lane & 15;

    // 64x64 WG tile; 4 waves in 2x2, each wave does 32x32 (2x2 frags of 16x16)
    const int m0 = blockIdx.x * 64 + (wave & 1) * 32;
    const int n0 = blockIdx.y * 64 + (wave >> 1) * 32;

    v8f acc[2][2] = {};

    for (int k0 = 0; k0 < K; k0 += 32) {
        v16bf af[2], bfm[2];
#pragma unroll
        for (int mi = 0; mi < 2; ++mi) {
            int row = m0 + mi * 16 + l16;
            if (row >= M) row = M - 1;                 // clamp (finite values in pad rows)
            af[mi] = make_a_frag(A + (long)row * lda + k0 + half * 16);
        }
#pragma unroll
        for (int ni = 0; ni < 2; ++ni) {
            int col = n0 + ni * 16 + l16;              // row of W
            bfm[ni] = *(const v16bf*)(W + (long)col * ldw + k0 + half * 16);
        }
#pragma unroll
        for (int mi = 0; mi < 2; ++mi)
#pragma unroll
            for (int ni = 0; ni < 2; ++ni)
                acc[mi][ni] = __builtin_amdgcn_wmma_f32_16x16x32_bf16(
                    false, af[mi], false, bfm[ni], (short)0, acc[mi][ni], false, false);
    }

#pragma unroll
    for (int mi = 0; mi < 2; ++mi) {
#pragma unroll
        for (int ni = 0; ni < 2; ++ni) {
            int ncol = n0 + ni * 16 + l16;
            float bv = bias[ncol];
            if (OUT_MODE == 2) {
                int mbase = m0 + mi * 16 + half * 8;   // 8 contiguous m per lane
                v8bf pk;
#pragma unroll
                for (int r = 0; r < 8; ++r) pk[r] = (bf16_t)(acc[mi][ni][r] + bv);
                *(v8bf*)((bf16_t*)Cout + (long)ncol * ldc + mbase) = pk;
            } else {
#pragma unroll
                for (int r = 0; r < 8; ++r) {
                    int m = m0 + mi * 16 + r + half * 8;
                    ((bf16_t*)Cout)[(long)m * ldc + ncol] = (bf16_t)(acc[mi][ni][r] + bv);
                }
            }
        }
    }
}

// One WG per (64 query rows, head). Q[4096][1024], Kb[1024][1024] (rows>=S finite pad),
// Vt[e_global][1024], output Obf[(b*8+h)][128] bf16.
__global__ __launch_bounds__(128) void attn_kernel(
    const bf16_t* __restrict__ Q,
    const bf16_t* __restrict__ Kb,
    const bf16_t* __restrict__ Vt,
    bf16_t* __restrict__ Obf,
    int S)
{
    extern __shared__ float lds[];                     // 64*SROWF scores + 256 reduction
    const int lane  = threadIdx.x & 31;
    const int wave  = threadIdx.x >> 5;
    const int half  = lane >> 4;
    const int l16   = lane & 15;
    const int bBase = blockIdx.x * 64;
    const int h     = blockIdx.y;
    const int hB    = h * 128;
    const int r0    = wave * 16;
    const float scale = 0.088388347648318447f;         // 1/sqrt(128)

    // ---- Phase 1: scores = Q @ K^T (E=128 -> 4 ksteps per 16-col tile) ----
    v16bf qf[4];
#pragma unroll
    for (int kk = 0; kk < 4; ++kk)
        qf[kk] = make_a_frag(Q + (long)(bBase + r0 + l16) * 1024 + hB + kk * 32 + half * 16);

    for (int st = 0; st < 64; ++st) {
        int s0 = st * 16;
        v8f c = {};
#pragma unroll
        for (int kk = 0; kk < 4; ++kk) {
            v16bf b = *(const v16bf*)(Kb + (long)(s0 + l16) * 1024 + hB + kk * 32 + half * 16);
            c = __builtin_amdgcn_wmma_f32_16x16x32_bf16(false, qf[kk], false, b, (short)0, c, false, false);
        }
#pragma unroll
        for (int r = 0; r < 8; ++r)
            lds[(r0 + r + half * 8) * SROWF + s0 + l16] = c[r] * scale;
    }
    __syncthreads();

    // ---- Softmax: 2 threads per row (halves [0,512) and [512,S)), LDS reduction ----
    {
        float* red = lds + 64 * SROWF;                 // 128 maxes + 128 sums
        const int r  = threadIdx.x & 63;
        const int hh = threadIdx.x >> 6;
        float* row = lds + r * SROWF;
        const int sbeg = hh * 512;
        const int send = hh ? S : 512;

        float mx = -1e30f;
        for (int s = sbeg; s < send; ++s) mx = fmaxf(mx, row[s]);
        red[threadIdx.x] = mx;
        __syncthreads();
        mx = fmaxf(red[r], red[r + 64]);

        float sum = 0.0f;
        for (int s = sbeg; s < send; ++s) sum += __expf(row[s] - mx);
        red[128 + threadIdx.x] = sum;
        __syncthreads();
        float inv = 1.0f / (red[128 + r] + red[128 + r + 64]);

        // In-place narrowing: half 0 -> bf16 elems [0,512) (bytes [0,1024));
        // half 1 -> bf16 elems [1024,1536) (bytes [2048,3072)). Each thread's writes
        // stay behind its own f32 reads and never touch the other thread's region.
        volatile float* vrow = row;
        volatile unsigned short* pb = (volatile unsigned short*)row;
        if (hh == 0) {
            for (int s = 0; s < 512; ++s) {
                float pv = __expf(vrow[s] - mx) * inv;
                union { bf16_t b; unsigned short u; } cv; cv.b = (bf16_t)pv;
                pb[s] = cv.u;
            }
        } else {
            for (int s = 512; s < S; ++s) {
                float pv = __expf(vrow[s] - mx) * inv;
                union { bf16_t b; unsigned short u; } cv; cv.b = (bf16_t)pv;
                pb[512 + s] = cv.u;
            }
            for (int s = S; s < 1024; ++s) pb[512 + s] = 0;  // zero padded probs
        }
    }
    __syncthreads();

    // ---- Phase 2: out = P @ V  (K-dim = 1024 -> 32 ksteps, 8 col-frags) ----
    v8f oacc[8] = {};
    const bf16_t* prow = (const bf16_t*)(lds + (r0 + l16) * SROWF);
    for (int s0 = 0; s0 < 1024; s0 += 32) {
        int sb = s0 + (s0 >= 512 ? 512 : 0);           // relocated second half
        v16bf a = make_a_frag(prow + sb + half * 16);   // ds_load_b128 x2
#pragma unroll
        for (int nf = 0; nf < 8; ++nf) {
            v16bf b = *(const v16bf*)(Vt + (long)(hB + nf * 16 + l16) * 1024 + s0 + half * 16);
            oacc[nf] = __builtin_amdgcn_wmma_f32_16x16x32_bf16(false, a, false, b, (short)0, oacc[nf], false, false);
        }
    }
#pragma unroll
    for (int nf = 0; nf < 8; ++nf)
#pragma unroll
        for (int r = 0; r < 8; ++r) {
            long orow = (long)(bBase + r0 + r + half * 8) * 8 + h;
            Obf[orow * 128 + nf * 16 + l16] = (bf16_t)oacc[nf][r];
        }
}

// out = A[32768x128] @ W[4096x128]^T + bias -> f32. W tile staged to LDS via TDM.
__global__ __launch_bounds__(128) void out_proj_kernel(
    const bf16_t* __restrict__ A,
    const bf16_t* __restrict__ W,
    const float* __restrict__ bias,
    float* __restrict__ C)
{
    __shared__ bf16_t wtile[64 * 128];                 // 16 KB: W rows nblk..nblk+63

    const int lane = threadIdx.x & 31;
    const int wave = threadIdx.x >> 5;
    const int half = lane >> 4;
    const int l16  = lane & 15;
    const int m0   = blockIdx.x * 64 + (wave & 1) * 32;
    const int nblk = blockIdx.y * 64;
    const int wn   = (wave >> 1) * 32;

    // TDM stage: wave 0 issues one tensor_load (2D: 128x64 bf16, contiguous rows)
    if (threadIdx.x < 32) {
        unsigned long long ga = (unsigned long long)(const void*)(W + (size_t)nblk * 128);
        unsigned lds_addr = (unsigned)(unsigned long long)(const void*)&wtile[0];
        u32x4 g0;
        g0[0] = 1u;                                        // count=1, load descriptor
        g0[1] = lds_addr;                                  // LDS byte address
        g0[2] = (unsigned)ga;                              // global_addr[31:0]
        g0[3] = ((unsigned)(ga >> 32) & 0x01FFFFFFu) | (2u << 30);  // addr[56:32] | type=2
        i32x8 g1;
        g1[0] = 1 << 16;                                   // data_size=1 -> 2-byte elems
        g1[1] = 128 << 16;                                 // tensor_dim0 = 128
        g1[2] = 64 << 16;                                  // tensor_dim1 = 64
        g1[3] = 128 << 16;                                 // tile_dim0 = 128
        g1[4] = 64;                                        // tile_dim1 = 64 (tile_dim2=0)
        g1[5] = 128;                                       // tensor_dim0_stride = 128
        g1[6] = 0;
        g1[7] = 0;
        i32x4 gz4 = {0, 0, 0, 0};
        i32x8 gz8 = {0, 0, 0, 0, 0, 0, 0, 0};
        __builtin_amdgcn_tensor_load_to_lds(g0, g1, gz4, gz4, gz8, 0);
        __builtin_amdgcn_s_wait_tensorcnt(0);
    }
    __syncthreads();

    v8f acc[2][2] = {};
#pragma unroll
    for (int k0 = 0; k0 < 128; k0 += 32) {
        v16bf af[2], bfm[2];
#pragma unroll
        for (int mi = 0; mi < 2; ++mi)
            af[mi] = make_a_frag(A + (size_t)(m0 + mi * 16 + l16) * 128 + k0 + half * 16);
#pragma unroll
        for (int ni = 0; ni < 2; ++ni)
            bfm[ni] = *(const v16bf*)(wtile + (wn + ni * 16 + l16) * 128 + k0 + half * 16);
#pragma unroll
        for (int mi = 0; mi < 2; ++mi)
#pragma unroll
            for (int ni = 0; ni < 2; ++ni)
                acc[mi][ni] = __builtin_amdgcn_wmma_f32_16x16x32_bf16(
                    false, af[mi], false, bfm[ni], (short)0, acc[mi][ni], false, false);
    }
#pragma unroll
    for (int mi = 0; mi < 2; ++mi)
#pragma unroll
        for (int ni = 0; ni < 2; ++ni) {
            int ncol = nblk + wn + ni * 16 + l16;
            float bv = bias[ncol];
#pragma unroll
            for (int r = 0; r < 8; ++r)
                C[(size_t)(m0 + mi * 16 + r + half * 8) * 4096 + ncol] = acc[mi][ni][r] + bv;
        }
}

extern "C" void kernel_launch(void* const* d_in, const int* in_sizes, int n_in,
                              void* d_out, int out_size, void* d_ws, size_t ws_size,
                              hipStream_t stream) {
    const float* target = (const float*)d_in[0];   // [4096,1024]
    const float* source = (const float*)d_in[1];   // [1000,4096]
    const float* value  = (const float*)d_in[2];   // [1000,4096]
    const float* Wq = (const float*)d_in[3];       // [1024,1024]
    const float* bq = (const float*)d_in[4];
    const float* Wk = (const float*)d_in[5];       // [1024,4096]
    const float* bk = (const float*)d_in[6];
    const float* Wv = (const float*)d_in[7];       // [1024,4096]
    const float* bv = (const float*)d_in[8];
    const float* Wo = (const float*)d_in[9];       // [4096,128]
    const float* bo = (const float*)d_in[10];
    float* out = (float*)d_out;                    // [32768,4096]

    const int B = 4096, S = 1000, DM = 1024, DL = 4096, H = 8, E = 128;
    const int Spad = 1024;

    bf16_t* p = (bf16_t*)d_ws;                     // ~63 MB of bf16 scratch
    bf16_t* tgt_bf = p; p += (size_t)B * DM;
    bf16_t* Wq_bf  = p; p += (size_t)DM * DM;
    bf16_t* src_bf = p; p += (size_t)S * DL;
    bf16_t* val_bf = p; p += (size_t)S * DL;
    bf16_t* Wk_bf  = p; p += (size_t)DM * DL;
    bf16_t* Wv_bf  = p; p += (size_t)DM * DL;
    bf16_t* Wo_bf  = p; p += (size_t)DL * E;
    bf16_t* q_bf   = p; p += (size_t)B * DM;
    bf16_t* k_bf   = p; p += (size_t)Spad * DM;    // padded rows (finite)
    bf16_t* vt_bf  = p; p += (size_t)DM * Spad;    // Vt[e_global][s]
    bf16_t* at_bf  = p; p += (size_t)B * H * E;

    auto cvt = [&](const float* s, bf16_t* d, long n) {
        cvt_f32_to_bf16<<<(int)(n / 8 / 256) + 1, 256, 0, stream>>>(s, d, n);
    };
    cvt(target, tgt_bf, (long)B * DM);
    cvt(Wq,     Wq_bf,  (long)DM * DM);
    cvt(source, src_bf, (long)S * DL);
    cvt(value,  val_bf, (long)S * DL);
    cvt(Wk,     Wk_bf,  (long)DM * DL);
    cvt(Wv,     Wv_bf,  (long)DM * DL);
    cvt(Wo,     Wo_bf,  (long)DL * E);

    dim3 blk(128);
    // q = target @ Wq^T + bq  -> bf16 [4096][1024]
    gemm_xwT<1><<<dim3(B / 64, DM / 64), blk, 0, stream>>>(tgt_bf, DM, Wq_bf, DM, bq, q_bf, DM, B, DM, DM);
    // k = source @ Wk^T + bk  -> bf16 [1024pad][1024]
    gemm_xwT<1><<<dim3((S + 63) / 64, DM / 64), blk, 0, stream>>>(src_bf, DL, Wk_bf, DL, bk, k_bf, DM, S, DM, DL);
    // vT = (value @ Wv^T + bv)^T -> bf16 Vt[1024][1024pad]
    gemm_xwT<2><<<dim3((S + 63) / 64, DM / 64), blk, 0, stream>>>(val_bf, DL, Wv_bf, DL, bv, vt_bf, Spad, S, DM, DL);

    // attention: one WG per (64 q-rows, head); ~259 KB dynamic LDS (CDNA5: 320 KB/WGP)
    size_t smem = ((size_t)64 * SROWF + 256) * sizeof(float);
    (void)hipFuncSetAttribute((const void*)attn_kernel,
                              hipFuncAttributeMaxDynamicSharedMemorySize, (int)smem);
    attn_kernel<<<dim3(B / 64, H), blk, smem, stream>>>(q_bf, k_bf, vt_bf, at_bf, S);

    // out = attn @ Wo^T + bo -> f32 [32768][4096], Wo tile staged via TDM
    out_proj_kernel<<<dim3(B * H / 64, DL / 64), blk, 0, stream>>>(at_bf, Wo_bf, bo, out);
}
